// ETypePromptModel_86706799772323
// MI455X (gfx1250) — matrix operand order repaired and checked by reading, not compiled, not run on hardware
//
#include <hip/hip_runtime.h>

// GEMM: out[16384,4096] = batchs[16384,256] * label2embed[4096,256]^T
// f32 inputs converted on the fly to f16 (v_cvt_pk_rtz_f16_f32),
// v_wmma_f32_16x16x32_f16 with f32 accumulation, f32 out.
// Memory-bound on the 256 MB output stream (~12 us at 23.3 TB/s); A+B (20 MB)
// stay resident in the 192 MB L2 so per-block panel re-reads are cheap.

typedef __attribute__((ext_vector_type(16))) _Float16 v16h;
typedef __attribute__((ext_vector_type(8)))  float    v8f;

#define GEMM_M 16384
#define GEMM_N 4096
#define GEMM_K 256     // L * D = 2 * 128
#define TILE_M 128
#define TILE_N 128
#define KCHUNK 64
#define LDS_STRIDE 72  // 64 f16 + 8 pad (144 B row stride, 16 B aligned)

union FragU {
    v16h  v;
    uint4 q[2];
};

// One VALU op: v_cvt_pk_rtz_f16_f32 packs two f32 -> two f16.
__device__ __forceinline__ unsigned int pack2_f16(float lo, float hi) {
    return __builtin_bit_cast(unsigned int, __builtin_amdgcn_cvt_pkrtz(lo, hi));
}

// A fragment 16x32 (MxK): lane<16 -> row M=lane, K = kk+[0..7] and kk+16+[0..7]
//                         lane>=16 -> row M=lane-16, K = kk+8+[0..7] and kk+24+[0..7]
__device__ __forceinline__ v16h load_frag_a(const unsigned short* s, int mt, int kk, int lane) {
    const int half = lane >> 4;        // 0 or 1
    const int mr   = lane & 15;
    const unsigned short* p = s + (mt + mr) * LDS_STRIDE + kk + half * 8;
    FragU u;
    u.q[0] = *reinterpret_cast<const uint4*>(p);        // K base+0..7   (16 B)
    u.q[1] = *reinterpret_cast<const uint4*>(p + 16);   // K base+16..23 (16 B)
    return u.v;
}

// B fragment 32x16 (KxN): lane<16 -> col N=lane,   K = kk+[0..7] and kk+16+[0..7]
//                         lane>=16 -> col N=lane-16, K = kk+8+[0..7] and kk+24+[0..7]
__device__ __forceinline__ v16h load_frag_b(const unsigned short* s, int nt, int kk, int lane) {
    const int half = lane >> 4;
    const int nr   = lane & 15;
    const unsigned short* p = s + (nt + nr) * LDS_STRIDE + kk + half * 8;
    FragU u;
    u.q[0] = *reinterpret_cast<const uint4*>(p);       // K base+0..7
    u.q[1] = *reinterpret_cast<const uint4*>(p + 16);  // K base+16..23
    return u.v;
}

__global__ __launch_bounds__(256)
void ETypePromptModel_gemm_wmma_f16(const float* __restrict__ A,   // [M, K]
                                    const float* __restrict__ Bm,  // [N, K]
                                    float* __restrict__ C) {       // [M, N]
    __shared__ unsigned short sA[TILE_M * LDS_STRIDE];  // 18 KB
    __shared__ unsigned short sB[TILE_N * LDS_STRIDE];  // 18 KB

    const int M0   = blockIdx.y * TILE_M;
    const int N0   = blockIdx.x * TILE_N;
    const int tid  = threadIdx.x;
    const int lane = tid & 31;
    const int wid  = tid >> 5;
    const int wm   = (wid & 3) * 32;   // wave's M offset in tile (4 waves in M)
    const int wn   = (wid >> 2) * 64;  // wave's N offset in tile (2 waves in N)

    v8f acc[2][4];
#pragma unroll
    for (int i = 0; i < 2; ++i)
#pragma unroll
        for (int j = 0; j < 4; ++j)
            acc[i][j] = (v8f){};

    for (int kc = 0; kc < GEMM_K; kc += KCHUNK) {
        __syncthreads();  // previous chunk fully consumed before overwrite

        // Stage A chunk: 128 rows x 64 k (f32 -> f16). 2048 float4 / 256 thr.
#pragma unroll
        for (int it = 0; it < (TILE_M * KCHUNK / 4) / 256; ++it) {
            const int i   = tid + it * 256;
            const int row = i >> 4;   // 16 float4 per row
            const int c4  = i & 15;
            const float4 v = *reinterpret_cast<const float4*>(
                A + (size_t)(M0 + row) * GEMM_K + kc + c4 * 4);
            uint2 p;
            p.x = pack2_f16(v.x, v.y);
            p.y = pack2_f16(v.z, v.w);
            *reinterpret_cast<uint2*>(&sA[row * LDS_STRIDE + c4 * 4]) = p;
        }
        // Stage B chunk: 128 cols x 64 k.
#pragma unroll
        for (int it = 0; it < (TILE_N * KCHUNK / 4) / 256; ++it) {
            const int i   = tid + it * 256;
            const int row = i >> 4;
            const int c4  = i & 15;
            const float4 v = *reinterpret_cast<const float4*>(
                Bm + (size_t)(N0 + row) * GEMM_K + kc + c4 * 4);
            uint2 p;
            p.x = pack2_f16(v.x, v.y);
            p.y = pack2_f16(v.z, v.w);
            *reinterpret_cast<uint2*>(&sB[row * LDS_STRIDE + c4 * 4]) = p;
        }
        __syncthreads();

        // Two k-steps of 32 within the chunk.
#pragma unroll
        for (int ks = 0; ks < KCHUNK; ks += 32) {
            v16h af[2], bfr[4];
#pragma unroll
            for (int i = 0; i < 2; ++i)
                af[i] = load_frag_a(sA, wm + i * 16, ks, lane);
#pragma unroll
            for (int j = 0; j < 4; ++j)
                bfr[j] = load_frag_b(sB, wn + j * 16, ks, lane);
#pragma unroll
            for (int i = 0; i < 2; ++i)
#pragma unroll
                for (int j = 0; j < 4; ++j)
                    acc[i][j] = __builtin_amdgcn_wmma_f32_16x16x32_f16(
                        /*neg_a=*/false, af[i], /*neg_b=*/false, bfr[j],
                        /*c_mod=*/(short)0, acc[i][j],
                        /*reuse_a=*/false, /*reuse_b=*/false);
        }
    }

    // Epilogue. C 16x16 layout: lanes 0-15 -> N=lane, VGPR r -> M=r;
    //                           lanes 16-31 -> N=lane-16, VGPR r -> M=8+r.
    const int mbase = M0 + wm + ((lane >> 4) << 3);
    const int nbase = N0 + wn + (lane & 15);
#pragma unroll
    for (int i = 0; i < 2; ++i)
#pragma unroll
        for (int j = 0; j < 4; ++j) {
            float* p = C + (size_t)(mbase + i * 16) * GEMM_N + nbase + j * 16;
#pragma unroll
            for (int r = 0; r < 8; ++r)
                p[(size_t)r * GEMM_N] = acc[i][j][r];
        }
}

extern "C" void kernel_launch(void* const* d_in, const int* in_sizes, int n_in,
                              void* d_out, int out_size, void* d_ws, size_t ws_size,
                              hipStream_t stream) {
    (void)in_sizes; (void)n_in; (void)out_size; (void)d_ws; (void)ws_size;
    const float* batchs      = (const float*)d_in[0];  // [16384, 2, 128] f32
    const float* label2embed = (const float*)d_in[1];  // [4096, 2, 128] f32
    float* out = (float*)d_out;                        // [16384, 4096] f32

    dim3 grid(GEMM_N / TILE_N, GEMM_M / TILE_M);  // (32, 128)
    dim3 block(256);                               // 8 waves (wave32)
    hipLaunchKernelGGL(ETypePromptModel_gemm_wmma_f16, grid, block, 0, stream,
                       batchs, label2embed, out);
}
